// SecondaryStructureEnergy_2628519985584
// MI455X (gfx1250) — matrix operand order repaired and checked by reading, not compiled, not run on hardware
//
#include <hip/hip_runtime.h>
#include <hip/hip_bf16.h>

typedef __attribute__((ext_vector_type(16))) _Float16 v16h;
typedef __attribute__((ext_vector_type(8)))  _Float16 v8h;
typedef __attribute__((ext_vector_type(8)))  float    v8f;

#define LSEQ  4096
#define BATCH 64
#define TPB   256            /* tiles per batch: 4096/16 */
#define W1F_HALFS (3*8*32*16)   /* 12288 halfs per MLP */
#define W2F_HALFS (4*8*32*16)   /* 16384 halfs per MLP */

struct F3 { float x, y, z; };
__device__ __forceinline__ F3 f3sub(F3 a, F3 b){ return {a.x-b.x, a.y-b.y, a.z-b.z}; }
__device__ __forceinline__ F3 f3cross(F3 a, F3 b){
  return {a.y*b.z - a.z*b.y, a.z*b.x - a.x*b.z, a.x*b.y - a.y*b.x};
}
__device__ __forceinline__ float f3dot(F3 a, F3 b){ return a.x*b.x + a.y*b.y + a.z*b.z; }

__device__ __forceinline__ void torsion_sc(F3 p0, F3 p1, F3 p2, F3 p3, float* s, float* c){
  F3 b1 = f3sub(p1, p0), b2 = f3sub(p2, p1), b3 = f3sub(p3, p2);
  F3 n1 = f3cross(b1, b2), n2 = f3cross(b2, b3);
  float bl = sqrtf(f3dot(b2, b2)) + 1e-8f;
  F3 b2n = {b2.x/bl, b2.y/bl, b2.z/bl};
  float x = f3dot(n1, n2);
  float y = f3dot(f3cross(n1, n2), b2n);
  float r = sqrtf(x*x + y*y) + 1e-20f;
  *s = y / r;  *c = x / r;
}

/* Load 16-bit A-fragment (16x32 MxK) from row-major LDS tile.
   Lane L: M = L&15; h[0..7] = K = kt*32 + (L>>4)*8 + j; h[8..15] = +16. */
__device__ __forceinline__ v16h load_a(const _Float16* base, int lo, int stride, int kt, int hi){
  const _Float16* p = base + lo*stride + kt*32 + hi*8;
  v8h a = *(const v8h*)p;
  v8h b = *(const v8h*)(p + 16);
  return __builtin_shufflevector(a, b, 0,1,2,3,4,5,6,7,8,9,10,11,12,13,14,15);
}

/* ---- Repack W1 (67/68x128, zero-pad K->96) and W2 (128x128) into per-lane
   fragment-linear fp16 arrays matching the 16-bit B VGPR layout:
   lane: N = nt*16 + (lane&15); h[j] = W[K = kt*32 + (lane>>4)*8 + (j&7) + (j>>3)*16][N]. ---- */
__global__ __launch_bounds__(256)
void prep_kernel(const float* __restrict__ tW1, const float* __restrict__ pW1,
                 const float* __restrict__ tW2, const float* __restrict__ pW2,
                 _Float16* __restrict__ w1f, _Float16* __restrict__ w2f)
{
  int gid = blockIdx.x * 256 + threadIdx.x;
  if (gid < 1536) {                      /* layer-1 fragments: 2 * 3 * 8 * 32 */
    int m  = gid / 768;
    int r  = gid % 768;
    int kt = r / 256;
    int r2 = r % 256;
    int nt = r2 / 32;
    int lane = r2 % 32;
    const float* W = m ? pW1 : tW1;
    int IN = m ? 68 : 67;
    _Float16* dst = w1f + m*W1F_HALFS + ((kt*8 + nt)*32 + lane)*16;
    int n  = nt*16 + (lane & 15);
    int kb = kt*32 + (lane >> 4)*8;
    #pragma unroll
    for (int j = 0; j < 16; j++) {
      int K = kb + (j & 7) + ((j >> 3) << 4);
      dst[j] = (K < IN) ? (_Float16)W[K*128 + n] : (_Float16)0.0f;
    }
  } else if (gid < 3584) {               /* layer-2 fragments: 2 * 4 * 8 * 32 */
    int g  = gid - 1536;
    int m  = g / 1024;
    int r  = g % 1024;
    int kt = r / 256;
    int r2 = r % 256;
    int nt = r2 / 32;
    int lane = r2 % 32;
    const float* W = m ? pW2 : tW2;
    _Float16* dst = w2f + m*W2F_HALFS + ((kt*8 + nt)*32 + lane)*16;
    int n  = nt*16 + (lane & 15);
    int kb = kt*32 + (lane >> 4)*8;
    #pragma unroll
    for (int j = 0; j < 16; j++)
      dst[j] = (_Float16)W[(kb + (j & 7) + ((j >> 3) << 4))*128 + n];
  }
}

/* ---- Fused featurize + MLP1 + MLP2 per 16-row tile, one tile per wave. ---- */
__global__ __launch_bounds__(128)
void energy_kernel(const float* __restrict__ R, const int* __restrict__ seq,
                   const float* __restrict__ emb,
                   const float* __restrict__ tb1, const float* __restrict__ tb2,
                   const float* __restrict__ tw3, const float* __restrict__ tb3,
                   const float* __restrict__ pb1, const float* __restrict__ pb2,
                   const float* __restrict__ pw3, const float* __restrict__ pb3,
                   const _Float16* __restrict__ w1f, const _Float16* __restrict__ w2f,
                   float* __restrict__ partials)
{
  __shared__ _Float16 xbuf[4][16*96];    /* per-wave feature tile (K padded to 96) */
  __shared__ _Float16 hbuf[4][16*128];   /* per-wave hidden tile                   */

  const int tid  = threadIdx.x;
  const int wv   = tid >> 5;
  const int lane = tid & 31;
  const int lo   = lane & 15;
  const int hi   = lane >> 4;
  const int t    = blockIdx.x * 4 + wv;  /* global tile id, 16384 total */
  const int b    = t >> 8;
  const int i0   = (t & (TPB - 1)) << 4;
  const int i    = i0 + lo;              /* row this lane featurizes */

  _Float16* xb = xbuf[wv];
  _Float16* hb = hbuf[wv];

  /* zero the K=67..95 pad tail once (avoid Inf/NaN bit-garbage * 0 weights) */
  if (hi == 0) {
    #pragma unroll
    for (int k = 67; k < 96; k++) xb[lo*96 + k] = (_Float16)0.0f;
  }

  /* ---- geometry for row i (clamped reads; invalid rows masked later) ---- */
  const long rb = (long)b * LSEQ;
  F3 P[5];
  #pragma unroll
  for (int k = 0; k < 5; k++) {
    int idx = i + k; if (idx > LSEQ - 1) idx = LSEQ - 1;
    const float* p = R + (rb + idx)*3;
    P[k] = {p[0], p[1], p[2]};
  }
  F3 u1 = f3sub(P[0], P[1]), u2 = f3sub(P[2], P[1]);
  F3 cr = f3cross(u1, u2);
  float dd  = f3dot(u1, u2);
  float cth = dd / sqrtf(dd*dd + f3dot(cr, cr) + 1e-30f);  /* cos(atan2(|c|,d)) */
  float s0, c0, s1, c1;
  torsion_sc(P[0], P[1], P[2], P[3], &s0, &c0);
  torsion_sc(P[1], P[2], P[3], P[4], &s1, &c1);

  /* ---- context embeddings e[i..i+3]: each half-wave part loads 2 rows ---- */
  float ef[32];
  {
    int ia = i + 2*hi;  if (ia > LSEQ - 1) ia = LSEQ - 1;
    int ib = ia + 1;    if (ib > LSEQ - 1) ib = LSEQ - 1;
    const float* ea = emb + seq[rb + ia]*16;
    const float* eb = emb + seq[rb + ib]*16;
    #pragma unroll
    for (int k = 0; k < 16; k++) { ef[k] = ea[k]; ef[16 + k] = eb[k]; }
  }

  float waveE = 0.0f;

  #pragma unroll
  for (int m = 0; m < 2; m++) {
    /* ---- featurize into xbuf (same-wave DS ops are in-order) ---- */
    __builtin_amdgcn_wave_barrier();
    {
      _Float16* xr = xb + lo*96;
      int co;
      if (m == 0) {
        co = 3;
        if (hi == 0) { xr[0] = (_Float16)cth; xr[1] = (_Float16)s0; xr[2] = (_Float16)c0; }
      } else {
        co = 4;
        if (hi == 0) { xr[0] = (_Float16)s0; xr[1] = (_Float16)c0;
                       xr[2] = (_Float16)s1; xr[3] = (_Float16)c1; }
      }
      #pragma unroll
      for (int k = 0; k < 32; k++) xr[co + 32*hi + k] = (_Float16)ef[k];
    }
    __builtin_amdgcn_wave_barrier();

    const _Float16* W1 = w1f + m*W1F_HALFS;
    const _Float16* W2 = w2f + m*W2F_HALFS;
    const float* B1 = m ? pb1 : tb1;
    const float* B2 = m ? pb2 : tb2;
    const float* W3 = m ? pw3 : tw3;
    const float  B3 = m ? pb3[0] : tb3[0];

    /* ---- layer 1: (16x96) @ (96x128), bias + ReLU -> hbuf fp16 ---- */
    v16h a0 = load_a(xb, lo, 96, 0, hi);
    v16h a1 = load_a(xb, lo, 96, 1, hi);
    v16h a2 = load_a(xb, lo, 96, 2, hi);
    #pragma unroll
    for (int nt = 0; nt < 8; nt++) {
      v8f c = {};
      c = __builtin_amdgcn_wmma_f32_16x16x32_f16(false, a0, false,
            *(const v16h*)(W1 + ((0*8 + nt)*32 + lane)*16), (short)0, c, false, false);
      c = __builtin_amdgcn_wmma_f32_16x16x32_f16(false, a1, false,
            *(const v16h*)(W1 + ((1*8 + nt)*32 + lane)*16), (short)0, c, false, false);
      c = __builtin_amdgcn_wmma_f32_16x16x32_f16(false, a2, false,
            *(const v16h*)(W1 + ((2*8 + nt)*32 + lane)*16), (short)0, c, false, false);
      float bias = B1[nt*16 + lo];
      #pragma unroll
      for (int r = 0; r < 8; r++) {
        float h = c[r] + bias;
        h = h > 0.0f ? h : 0.0f;
        hb[(r + 8*hi)*128 + nt*16 + lo] = (_Float16)h;  /* D->A lane transpose via LDS */
      }
    }
    __builtin_amdgcn_wave_barrier();

    /* ---- layer 2: (16x128) @ (128x128), bias + ReLU, fused dot with W3 ---- */
    v16h h0 = load_a(hb, lo, 128, 0, hi);
    v16h h1 = load_a(hb, lo, 128, 1, hi);
    v16h h2 = load_a(hb, lo, 128, 2, hi);
    v16h h3 = load_a(hb, lo, 128, 3, hi);
    float rowE[8] = {0,0,0,0,0,0,0,0};
    #pragma unroll
    for (int nt = 0; nt < 8; nt++) {
      v8f c = {};
      c = __builtin_amdgcn_wmma_f32_16x16x32_f16(false, h0, false,
            *(const v16h*)(W2 + ((0*8 + nt)*32 + lane)*16), (short)0, c, false, false);
      c = __builtin_amdgcn_wmma_f32_16x16x32_f16(false, h1, false,
            *(const v16h*)(W2 + ((1*8 + nt)*32 + lane)*16), (short)0, c, false, false);
      c = __builtin_amdgcn_wmma_f32_16x16x32_f16(false, h2, false,
            *(const v16h*)(W2 + ((2*8 + nt)*32 + lane)*16), (short)0, c, false, false);
      c = __builtin_amdgcn_wmma_f32_16x16x32_f16(false, h3, false,
            *(const v16h*)(W2 + ((3*8 + nt)*32 + lane)*16), (short)0, c, false, false);
      float bias = B2[nt*16 + lo];
      float w3v  = W3[nt*16 + lo];
      #pragma unroll
      for (int r = 0; r < 8; r++) {
        float h = c[r] + bias;
        h = h > 0.0f ? h : 0.0f;
        rowE[r] += h * w3v;
      }
    }

    /* per-row sum over N: reduce across the 16 lanes of each half-wave */
    #pragma unroll
    for (int r = 0; r < 8; r++) {
      float v = rowE[r];
      v += __shfl_xor(v, 1, 32);
      v += __shfl_xor(v, 2, 32);
      v += __shfl_xor(v, 4, 32);
      v += __shfl_xor(v, 8, 32);
      rowE[r] = v;   /* uniform within each 16-lane group */
    }
    const int npos = (m == 0) ? (LSEQ - 3) : (LSEQ - 4);
    float e = 0.0f;
    #pragma unroll
    for (int r = 0; r < 8; r++) {
      int row = i0 + r + 8*hi;
      if (row < npos) e += rowE[r] + B3;   /* mask padded rows, +b3 per valid row */
    }
    e += __shfl_xor(e, 16, 32);            /* combine the two half-wave groups */
    waveE += e;
  }

  if (lane == 0) partials[t] = waveE;
}

/* ---- deterministic per-batch reduction of the 256 tile partials ---- */
__global__ __launch_bounds__(256)
void reduce_kernel(const float* __restrict__ partials, float* __restrict__ out)
{
  __shared__ float sbuf[8];
  int b = blockIdx.x, tid = threadIdx.x;
  float v = partials[b*TPB + tid];
  v += __shfl_xor(v, 1, 32);
  v += __shfl_xor(v, 2, 32);
  v += __shfl_xor(v, 4, 32);
  v += __shfl_xor(v, 8, 32);
  v += __shfl_xor(v, 16, 32);
  if ((tid & 31) == 0) sbuf[tid >> 5] = v;
  __syncthreads();
  if (tid == 0) {
    float e = 0.0f;
    #pragma unroll
    for (int k = 0; k < 8; k++) e += sbuf[k];
    out[b] = e;
  }
}

extern "C" void kernel_launch(void* const* d_in, const int* in_sizes, int n_in,
                              void* d_out, int out_size, void* d_ws, size_t ws_size,
                              hipStream_t stream)
{
  const float* R   = (const float*)d_in[0];
  const int*   seq = (const int*)  d_in[1];
  const float* emb = (const float*)d_in[2];
  const float* tW1 = (const float*)d_in[3];
  const float* tb1 = (const float*)d_in[4];
  const float* tW2 = (const float*)d_in[5];
  const float* tb2 = (const float*)d_in[6];
  const float* tW3 = (const float*)d_in[7];
  const float* tb3 = (const float*)d_in[8];
  const float* pW1 = (const float*)d_in[9];
  const float* pb1 = (const float*)d_in[10];
  const float* pW2 = (const float*)d_in[11];
  const float* pb2 = (const float*)d_in[12];
  const float* pW3 = (const float*)d_in[13];
  const float* pb3 = (const float*)d_in[14];

  char* ws = (char*)d_ws;
  _Float16* w1f     = (_Float16*)(ws);                    /* 2*12288 halfs = 48 KB */
  _Float16* w2f     = (_Float16*)(ws + 49152);            /* 2*16384 halfs = 64 KB */
  float*    partials = (float*)(ws + 49152 + 65536);      /* 16384 f32     = 64 KB */
  float*    out     = (float*)d_out;

  hipLaunchKernelGGL(prep_kernel, dim3(14), dim3(256), 0, stream,
                     tW1, pW1, tW2, pW2, w1f, w2f);
  hipLaunchKernelGGL(energy_kernel, dim3(BATCH * TPB / 4), dim3(128), 0, stream,
                     R, seq, emb, tb1, tb2, tW3, tb3, pb1, pb2, pW3, pb3,
                     w1f, w2f, partials);
  hipLaunchKernelGGL(reduce_kernel, dim3(BATCH), dim3(256), 0, stream, partials, out);
}
